// MaskedCrossAttention_56805237457561
// MI455X (gfx1250) — matrix operand order repaired and checked by reading, not compiled, not run on hardware
//
#include <hip/hip_runtime.h>
#include <hip/hip_bf16.h>

typedef __attribute__((ext_vector_type(16))) _Float16     v16h;
typedef __attribute__((ext_vector_type(8)))  float        v8f;
typedef __attribute__((ext_vector_type(4)))  unsigned int u32x4;

#define HEADS    8
#define DIM_HEAD 64
#define DIMX     1024
#define INNER    512      // HEADS * DIM_HEAD
#define NQ       2048
#define MK       8224
#define MASK_BIG 1.0e9f

// ---------------------------------------------------------------------------
// Fragment loaders (CDNA5 ISA §7.12.2), vectorized to 2x b128/lane.
// ---------------------------------------------------------------------------

// A-matrix 16x32 f16, row-major source (ld in halves, multiple of 8).
// lane L: row M=L%16; VGPRs 0..3 hold K = hi*8 + 0..7, VGPRs 4..7 K = 16+hi*8+0..7.
__device__ __forceinline__ v16h load_a_frag(const _Float16* __restrict__ base,
                                            size_t ld, int lane) {
  const int hi = (lane >> 4) & 1;
  const int m  = lane & 15;
  union { v16h h; u32x4 q[2]; } f;
  const _Float16* p = base + (size_t)m * ld + hi * 8;
  f.q[0] = *reinterpret_cast<const u32x4*>(p);
  f.q[1] = *reinterpret_cast<const u32x4*>(p + 16);
  return f.h;
}

// B-matrix 32x16 f16 where logical B[k][n] = base[n*ld + k] (K-contiguous
// source, i.e. transposed-weight / K^T / V^T storage). lane L: col N=L%16,
// K = hi*8 + 0..7 and 16 + hi*8 + 0..7 -> two b128 loads.
__device__ __forceinline__ v16h load_bT_frag(const _Float16* __restrict__ base,
                                             size_t ld, int lane) {
  const int hi = (lane >> 4) & 1;
  const int n  = lane & 15;
  union { v16h h; u32x4 q[2]; } f;
  const _Float16* p = base + (size_t)n * ld + hi * 8;
  f.q[0] = *reinterpret_cast<const u32x4*>(p);
  f.q[1] = *reinterpret_cast<const u32x4*>(p + 16);
  return f.h;
}

// 16-lane group reductions (C/D rows live across one 16-lane half)
__device__ __forceinline__ float rowmax16(float v) {
#pragma unroll
  for (int m = 1; m < 16; m <<= 1) v = fmaxf(v, __shfl_xor(v, m, 32));
  return v;
}
__device__ __forceinline__ float rowsum16(float v) {
#pragma unroll
  for (int m = 1; m < 16; m <<= 1) v += __shfl_xor(v, m, 32);
  return v;
}

// ---------------------------------------------------------------------------
// Stage 0: conversions. Plain for activations; transposing for weights so
// all B fragments become K-contiguous b128 loads.
// ---------------------------------------------------------------------------
__global__ void cvt_f32_to_f16(const float* __restrict__ in,
                               _Float16* __restrict__ out, int n) {
  int i = blockIdx.x * blockDim.x + threadIdx.x;
  if (i < n) out[i] = (_Float16)in[i];
}

// in: K x N (row-major f32) -> out: N x K (row-major f16) == W^T
__global__ void cvt_f32_to_f16_T(const float* __restrict__ in,
                                 _Float16* __restrict__ out, int K, int N) {
  int i = blockIdx.x * blockDim.x + threadIdx.x;
  if (i < K * N) {
    int k = i / N, n = i - k * N;
    out[(size_t)n * K + k] = (_Float16)in[i];
  }
}

// ---------------------------------------------------------------------------
// WMMA GEMM, 16x64 output per wave (4 col tiles reuse one A fragment).
//   C(Mr x Nc) = A(Mr x K) @ Bt^T, Bt stored Nc x K (transposed weights).
// f16 output with scale.
// ---------------------------------------------------------------------------
__global__ __launch_bounds__(256)
void gemm_wmma_f16out(const _Float16* __restrict__ A,
                      const _Float16* __restrict__ Bt,
                      _Float16* __restrict__ C,
                      int Mr, int Nc, int K, float scale) {
  const int lane = threadIdx.x & 31;
  const int wave = (blockIdx.x * blockDim.x + threadIdx.x) >> 5;
  const int ctiles = Nc >> 6;                 // 64-wide column strips
  const int row0 = (wave / ctiles) << 4;
  const int col0 = (wave % ctiles) << 6;
  if (row0 >= Mr) return;

  v8f acc[4] = {v8f{}, v8f{}, v8f{}, v8f{}};
  const _Float16* arow = A + (size_t)row0 * K;
  for (int k0 = 0; k0 < K; k0 += 32) {
    v16h a = load_a_frag(arow + k0, K, lane);
    __builtin_prefetch(arow + k0 + 32 + (size_t)(lane & 15) * K, 0, 0);
#pragma unroll
    for (int jt = 0; jt < 4; ++jt) {
      v16h b = load_bT_frag(Bt + (size_t)(col0 + jt * 16) * K + k0, K, lane);
      acc[jt] = __builtin_amdgcn_wmma_f32_16x16x32_f16(false, a, false, b,
                                                       (short)0, acc[jt], false, false);
    }
  }
  const int hi = (lane >> 4) & 1;
  const int n  = lane & 15;
#pragma unroll
  for (int r = 0; r < 8; ++r) {
    const size_t row = (size_t)(row0 + r + hi * 8);
#pragma unroll
    for (int jt = 0; jt < 4; ++jt)
      C[row * Nc + col0 + jt * 16 + n] = (_Float16)(acc[jt][r] * scale);
  }
}

// kv GEMM: Nc = 1024; cols [0,512) -> kh row-major (MK x 512);
// cols [512,1024) -> vt transposed (INNER x MK) so P@V loads are K-contiguous.
__global__ __launch_bounds__(256)
void gemm_wmma_kv(const _Float16* __restrict__ A,
                  const _Float16* __restrict__ Bt,
                  _Float16* __restrict__ kh,
                  _Float16* __restrict__ vt,
                  int Mr, int K) {
  const int Nc = 2 * INNER;
  const int lane = threadIdx.x & 31;
  const int wave = (blockIdx.x * blockDim.x + threadIdx.x) >> 5;
  const int ctiles = Nc >> 6;                 // 16
  const int row0 = (wave / ctiles) << 4;
  const int col0 = (wave % ctiles) << 6;
  if (row0 >= Mr) return;

  v8f acc[4] = {v8f{}, v8f{}, v8f{}, v8f{}};
  const _Float16* arow = A + (size_t)row0 * K;
  for (int k0 = 0; k0 < K; k0 += 32) {
    v16h a = load_a_frag(arow + k0, K, lane);
    __builtin_prefetch(arow + k0 + 32 + (size_t)(lane & 15) * K, 0, 0);
#pragma unroll
    for (int jt = 0; jt < 4; ++jt) {
      v16h b = load_bT_frag(Bt + (size_t)(col0 + jt * 16) * K + k0, K, lane);
      acc[jt] = __builtin_amdgcn_wmma_f32_16x16x32_f16(false, a, false, b,
                                                       (short)0, acc[jt], false, false);
    }
  }
  const int hi = (lane >> 4) & 1;
  const int n  = lane & 15;
#pragma unroll
  for (int r = 0; r < 8; ++r) {
    const size_t row = (size_t)(row0 + r + hi * 8);
#pragma unroll
    for (int jt = 0; jt < 4; ++jt) {
      const int col = col0 + jt * 16 + n;
      const _Float16 val = (_Float16)acc[jt][r];
      if (col < INNER) kh[row * INNER + col] = val;                 // K half
      else             vt[(size_t)(col - INNER) * MK + row] = val;  // V half, transposed
    }
  }
}

// Final projection: f32 output straight to d_out.
__global__ __launch_bounds__(256)
void gemm_wmma_f32out(const _Float16* __restrict__ A,
                      const _Float16* __restrict__ Bt,
                      float* __restrict__ C,
                      int Mr, int Nc, int K) {
  const int lane = threadIdx.x & 31;
  const int wave = (blockIdx.x * blockDim.x + threadIdx.x) >> 5;
  const int ctiles = Nc >> 6;
  const int row0 = (wave / ctiles) << 4;
  const int col0 = (wave % ctiles) << 6;
  if (row0 >= Mr) return;

  v8f acc[4] = {v8f{}, v8f{}, v8f{}, v8f{}};
  const _Float16* arow = A + (size_t)row0 * K;
  for (int k0 = 0; k0 < K; k0 += 32) {
    v16h a = load_a_frag(arow + k0, K, lane);
#pragma unroll
    for (int jt = 0; jt < 4; ++jt) {
      v16h b = load_bT_frag(Bt + (size_t)(col0 + jt * 16) * K + k0, K, lane);
      acc[jt] = __builtin_amdgcn_wmma_f32_16x16x32_f16(false, a, false, b,
                                                       (short)0, acc[jt], false, false);
    }
  }
  const int hi = (lane >> 4) & 1;
  const int n  = lane & 15;
#pragma unroll
  for (int r = 0; r < 8; ++r) {
    const size_t row = (size_t)(row0 + r + hi * 8);
#pragma unroll
    for (int jt = 0; jt < 4; ++jt)
      C[row * Nc + col0 + jt * 16 + n] = acc[jt][r];
  }
}

// ---------------------------------------------------------------------------
// Streaming-softmax attention. One wave per (head, 16-query tile), keys in
// chunks of 32. kh: MK x INNER row-major; vt: INNER x MK (d-major).
// ---------------------------------------------------------------------------
__global__ __launch_bounds__(256)
void attn_wmma(const _Float16* __restrict__ qh,   // NQ x INNER
               const _Float16* __restrict__ kh,   // MK x INNER
               const _Float16* __restrict__ vt,   // INNER x MK
               const float* __restrict__ mask,    // NQ x MK
               _Float16* __restrict__ oh) {       // NQ x INNER
  __shared__ _Float16 plds[8][16 * 32];

  const int lane = threadIdx.x & 31;
  const int wv   = threadIdx.x >> 5;
  const int task = blockIdx.x * 8 + wv;          // 0 .. 1023
  const int head = task >> 7;
  const int n0   = (task & 127) << 4;
  const int hi   = (lane >> 4) & 1;
  const int nn   = lane & 15;
  const int dofs = head * DIM_HEAD;

  const v16h aq0 = load_a_frag(qh + (size_t)n0 * INNER + dofs,      INNER, lane);
  const v16h aq1 = load_a_frag(qh + (size_t)n0 * INNER + dofs + 32, INNER, lane);

  v8f o[4] = {v8f{}, v8f{}, v8f{}, v8f{}};
  float mrun[8], lrun[8];
#pragma unroll
  for (int r = 0; r < 8; ++r) { mrun[r] = -3.0e38f; lrun[r] = 0.0f; }

  for (int mc = 0; mc < MK; mc += 32) {
    // ---- scores: two 16x16 tiles (key cols [mc,mc+16), [mc+16,mc+32))
    const _Float16* kbase = kh + (size_t)mc * INNER + dofs;
    v8f s0 = {}, s1 = {};
    {
      v16h bk = load_bT_frag(kbase, INNER, lane);
      s0 = __builtin_amdgcn_wmma_f32_16x16x32_f16(false, aq0, false, bk, (short)0, s0, false, false);
      bk = load_bT_frag(kbase + 32, INNER, lane);
      s0 = __builtin_amdgcn_wmma_f32_16x16x32_f16(false, aq1, false, bk, (short)0, s0, false, false);
      bk = load_bT_frag(kbase + (size_t)16 * INNER, INNER, lane);
      s1 = __builtin_amdgcn_wmma_f32_16x16x32_f16(false, aq0, false, bk, (short)0, s1, false, false);
      bk = load_bT_frag(kbase + (size_t)16 * INNER + 32, INNER, lane);
      s1 = __builtin_amdgcn_wmma_f32_16x16x32_f16(false, aq1, false, bk, (short)0, s1, false, false);
    }

    // ---- mask bias (streamed once: non-temporal) + online softmax
#pragma unroll
    for (int r = 0; r < 8; ++r) {
      const size_t row = (size_t)(n0 + r + hi * 8);
      s0[r] += __builtin_nontemporal_load(&mask[row * MK + mc + nn])      * MASK_BIG;
      s1[r] += __builtin_nontemporal_load(&mask[row * MK + mc + 16 + nn]) * MASK_BIG;

      float mx    = rowmax16(fmaxf(s0[r], s1[r]));
      float mnew  = fmaxf(mrun[r], mx);
      float alpha = __expf(mrun[r] - mnew);
      float p0 = __expf(s0[r] - mnew);
      float p1 = __expf(s1[r] - mnew);
      lrun[r] = lrun[r] * alpha + rowsum16(p0 + p1);
      mrun[r] = mnew;
#pragma unroll
      for (int j = 0; j < 4; ++j) o[j][r] *= alpha;
      plds[wv][(r + hi * 8) * 32 + nn]      = (_Float16)p0;
      plds[wv][(r + hi * 8) * 32 + 16 + nn] = (_Float16)p1;
    }

    // ---- O += P @ V  (LDS DS ops in-order within a wave; V^T is K-contiguous)
    v16h ap = load_a_frag(&plds[wv][0], 32, lane);
#pragma unroll
    for (int j = 0; j < 4; ++j) {
      v16h bv = load_bT_frag(vt + (size_t)(dofs + j * 16) * MK + mc, MK, lane);
      o[j] = __builtin_amdgcn_wmma_f32_16x16x32_f16(false, ap, false, bv,
                                                    (short)0, o[j], false, false);
    }
  }

  // ---- normalize, emit f16 for the out-projection
#pragma unroll
  for (int r = 0; r < 8; ++r) {
    const float inv = 1.0f / lrun[r];
    const size_t row = (size_t)(n0 + r + hi * 8);
#pragma unroll
    for (int j = 0; j < 4; ++j)
      oh[row * INNER + dofs + j * 16 + nn] = (_Float16)(o[j][r] * inv);
  }
}

// ---------------------------------------------------------------------------
// Host-side orchestration
// ---------------------------------------------------------------------------
extern "C" void kernel_launch(void* const* d_in, const int* in_sizes, int n_in,
                              void* d_out, int out_size, void* d_ws, size_t ws_size,
                              hipStream_t stream) {
  const float* x    = (const float*)d_in[0];   // NQ x DIMX
  const float* y    = (const float*)d_in[1];   // MK x DIMX
  const float* mask = (const float*)d_in[2];   // NQ x MK
  const float* W_q  = (const float*)d_in[3];   // DIMX x INNER
  const float* W_kv = (const float*)d_in[4];   // DIMX x 2*INNER
  const float* W_o  = (const float*)d_in[5];   // INNER x DIMX
  float* out = (float*)d_out;                  // NQ x DIMX

  _Float16* p = (_Float16*)d_ws;
  _Float16* xh   = p; p += (size_t)NQ * DIMX;
  _Float16* yh   = p; p += (size_t)MK * DIMX;
  _Float16* wqt  = p; p += (size_t)INNER * DIMX;        // W_q^T
  _Float16* wkvt = p; p += (size_t)2 * INNER * DIMX;    // W_kv^T
  _Float16* wot  = p; p += (size_t)DIMX * INNER;        // W_out^T
  _Float16* qh   = p; p += (size_t)NQ * INNER;
  _Float16* kh   = p; p += (size_t)MK * INNER;
  _Float16* vt   = p; p += (size_t)INNER * MK;
  _Float16* ohb  = p; p += (size_t)NQ * INNER;

  cvt_f32_to_f16<<<(NQ * DIMX + 255) / 256, 256, 0, stream>>>(x, xh, NQ * DIMX);
  cvt_f32_to_f16<<<(MK * DIMX + 255) / 256, 256, 0, stream>>>(y, yh, MK * DIMX);
  cvt_f32_to_f16_T<<<(DIMX * INNER + 255) / 256, 256, 0, stream>>>(W_q, wqt, DIMX, INNER);
  cvt_f32_to_f16_T<<<(DIMX * 2 * INNER + 255) / 256, 256, 0, stream>>>(W_kv, wkvt, DIMX, 2 * INNER);
  cvt_f32_to_f16_T<<<(INNER * DIMX + 255) / 256, 256, 0, stream>>>(W_o, wot, INNER, DIMX);

  // q = x @ W_q * dim_head^-0.5 : waves = 128 * (512/64) = 1024
  gemm_wmma_f16out<<<1024 / 8, 256, 0, stream>>>(xh, wqt, qh, NQ, INNER, DIMX, 0.125f);

  // kv = y @ W_kv -> kh + vt    : waves = 514 * (1024/64) = 8224
  gemm_wmma_kv<<<8224 / 8, 256, 0, stream>>>(yh, wkvt, kh, vt, MK, DIMX);

  // attention: 8 heads x 128 query tiles = 1024 wave-tasks
  attn_wmma<<<128, 256, 0, stream>>>(qh, kh, vt, mask, ohb);

  // out = o @ W_out : waves = 128 * (1024/64) = 2048
  gemm_wmma_f32out<<<2048 / 8, 256, 0, stream>>>(ohb, wot, out, NQ, DIMX, INNER);
}
